// GraphFeatureTokenizer_22737556865608
// MI455X (gfx1250) — matrix operand order repaired and checked by reading, not compiled, not run on hardware
//
#include <hip/hip_runtime.h>
#include <hip/hip_bf16.h>
#include <math.h>

typedef __attribute__((ext_vector_type(16))) _Float16 v16h;
typedef __attribute__((ext_vector_type(8)))  float    v8f;

#define N_NODESC   30000
#define DEGC       8
#define N_EDGESC   (N_NODESC * DEGC)          // 240000
#define EMBEDC     512
#define HDIMC      2048                        // [rbf FF | dir FF] = 1024 + 1024
#define NUM_GC     50
#define AKC        64                          // padded layer-1 K (50 rbf + 3 dir + pad)
#define PAIRSC     (N_NODESC * DEGC * DEGC)    // 1,920,000
#define TOTALEC    (3 * N_EDGESC + 2 * PAIRSC) // 4,560,000
#define RBF_RADIUSC 12.0f
#define INV_SQRT2C 0.70710678118654752440f
#define INV_SQRT3C 0.57735026918962576451f

__device__ __forceinline__ float gelu_tanh(float x) {
    const float k0 = 0.7978845608028654f;   // sqrt(2/pi)
    const float k1 = 0.044715f;
    float x3 = x * x * x;
    return 0.5f * x * (1.0f + tanhf(k0 * (x + k1 * x3)));
}

// ---- WMMA fragment helpers (wave32, this file's contiguous-k order k = 16*half + s for A and B) -----------------
// A 16x32 f16 from row-major LDS tile: lane m = lane&15, half = lane>>4.
// VGPR v (halves 2v,2v+1): K = half*16 + 2*v
// -> per lane: 16 contiguous halves at [m][half*16]
__device__ __forceinline__ v16h load_a_frag(const _Float16* base, int ld, int lane) {
    int m = lane & 15, half = lane >> 4;
    v16h a;
#pragma unroll
    for (int v = 0; v < 8; ++v) {
        int kk = half * 16 + 2 * v;
        a[2 * v]     = base[m * ld + kk];
        a[2 * v + 1] = base[m * ld + kk + 1];
    }
    return a;
}
// B fragment from pre-swizzled global weights: [tile][lane][16 halves]
__device__ __forceinline__ v16h load_frag32(const _Float16* base, int tile, int lane) {
    return ((const v16h*)base)[(size_t)tile * 32 + lane];
}

// ---- weight prep: swizzle into per-lane-contiguous fragment layout ------
// W1s: K=64 (2 k-blocks of 32) x 128 col-tiles of 16. tile = kb32*128 + ct.
// lane chunk element i -> B[kb32*32 + (lane>>4)*16 + i][ct*16 + (lane&15)]
__global__ void prep_w1_swz(const float* __restrict__ rbf_w1,  // [50,1024]
                            const float* __restrict__ dir_w1,  // [3,1024]
                            _Float16* __restrict__ W1s) {      // [2*128][32][16]
    int idx = blockIdx.x * blockDim.x + threadIdx.x;
    if (idx >= AKC * HDIMC) return;
    int tile = idx >> 9;           // /512
    int lane = (idx >> 4) & 31;
    int i    = idx & 15;
    int kb32 = tile >> 7;          // 0..1
    int ct   = tile & 127;         // 0..127
    int k = kb32 * 32 + ((lane >> 4) << 4) + i;   // 0..63
    int j = ct * 16 + (lane & 15);                // 0..2047
    float v = 0.0f;
    if (j < 1024) { if (k < NUM_GC) v = rbf_w1[k * 1024 + j]; }
    else          { if (k >= NUM_GC && k < NUM_GC + 3) v = dir_w1[(k - NUM_GC) * 1024 + (j - 1024)]; }
    W1s[idx] = (_Float16)v;
}

// W2s: K=2048 (64 k-blocks of 32) x 32 col-tiles of 16. tile = kt*32 + ct.
__global__ void prep_w2_swz(const float* __restrict__ rbf_w2,  // [1024,512]
                            const float* __restrict__ dir_w2,  // [1024,512]
                            _Float16* __restrict__ W2s) {      // [64*32][32][16]
    int idx = blockIdx.x * blockDim.x + threadIdx.x;
    if (idx >= HDIMC * EMBEDC) return;
    int tile = idx >> 9;
    int lane = (idx >> 4) & 31;
    int i    = idx & 15;
    int kt = tile >> 5;            // 0..63
    int ct = tile & 31;            // 0..31
    int k = kt * 32 + ((lane >> 4) << 4) + i;     // 0..2047
    int j = ct * 16 + (lane & 15);                // 0..511
    float v = (k < 1024) ? rbf_w2[k * EMBEDC + j] : dir_w2[(k - 1024) * EMBEDC + j];
    W2s[idx] = (_Float16)v;
}

__global__ void prep_b1_kernel(const float* __restrict__ rbf_b1,
                               const float* __restrict__ dir_b1,
                               float* __restrict__ b1c) {         // [2048]
    int j = blockIdx.x * blockDim.x + threadIdx.x;
    if (j >= HDIMC) return;
    b1c[j] = (j < 1024) ? rbf_b1[j] : dir_b1[j - 1024];
}

__global__ void prep_badd_kernel(const float* __restrict__ rbf_b2,
                                 const float* __restrict__ dir_b2,
                                 const float* __restrict__ type_emb,  // [2,512]
                                 float* __restrict__ badd) {          // [512]
    int j = blockIdx.x * blockDim.x + threadIdx.x;
    if (j >= EMBEDC) return;
    badd[j] = rbf_b2[j] + dir_b2[j] + type_emb[EMBEDC + j];
}

// ---- node tokens ---------------------------------------------------------
__global__ void node_tok_kernel(const float* __restrict__ anum_emb,
                                const float* __restrict__ type_emb,
                                const int* __restrict__ anum,
                                float* __restrict__ out_tokens) {
    int idx = blockIdx.x * blockDim.x + threadIdx.x;
    if (idx >= N_NODESC * EMBEDC) return;
    int i = idx / EMBEDC, j = idx % EMBEDC;
    int an = anum[i];
    out_tokens[idx] = INV_SQRT2C * (anum_emb[(size_t)an * EMBEDC + j] + type_emb[j]);
}

// ---- edge geometry -> packed f16 A matrix [E, 64] ------------------------
__global__ void edge_geom_kernel(const float* __restrict__ pos,
                                 const int* __restrict__ ei,
                                 _Float16* __restrict__ A1) {
    int e = blockIdx.x * blockDim.x + threadIdx.x;
    if (e >= N_EDGESC) return;
    int s = ei[e], d = ei[N_EDGESC + e];
    float vx = pos[3 * d + 0] - pos[3 * s + 0];
    float vy = pos[3 * d + 1] - pos[3 * s + 1];
    float vz = pos[3 * d + 2] - pos[3 * s + 2];
    float dist = sqrtf(vx * vx + vy * vy + vz * vz);
    float inv  = 1.0f / fmaxf(dist, 1e-12f);
    const float step  = RBF_RADIUSC / (float)(NUM_GC - 1);
    const float coeff = -0.5f / (step * step);
    __align__(16) _Float16 row[AKC];
#pragma unroll
    for (int g = 0; g < NUM_GC; ++g) {
        float t = dist - (float)g * step;
        row[g] = (_Float16)expf(coeff * t * t);
    }
    row[50] = (_Float16)(vx * inv);
    row[51] = (_Float16)(vy * inv);
    row[52] = (_Float16)(vz * inv);
#pragma unroll
    for (int g = 53; g < AKC; ++g) row[g] = (_Float16)0.0f;
    uint4* dst = (uint4*)(A1 + (size_t)e * AKC);
    const uint4* srcv = (const uint4*)row;
#pragma unroll
    for (int i = 0; i < 8; ++i) dst[i] = srcv[i];
}

// ---- fused 2-layer MLP: GEMM1 + GELU + GEMM2, all WMMA -------------------
// Block: 256 threads (8 waves), M-tile = 64 edges, N = 512 full.
__global__ __launch_bounds__(256)
void fused_edge_mlp(const _Float16* __restrict__ A1,   // [E, 64]
                    const _Float16* __restrict__ W1s,  // swizzled frags
                    const float*    __restrict__ b1c,  // [2048]
                    const _Float16* __restrict__ W2s,  // swizzled frags
                    const float*    __restrict__ badd, // [512]
                    float* __restrict__ out_tokens) {
    __shared__ __align__(32) _Float16 As[64 * AKC];    // 8 KB
    __shared__ __align__(32) _Float16 Hs[64 * 64];     // 8 KB (gelu'd H chunk)

    const int tid  = threadIdx.x;
    const int lane = tid & 31;
    const int w    = tid >> 5;                          // wave id 0..7
    const int e0   = blockIdx.x * 64;
    const int n    = lane & 15;
    const int half = lane >> 4;

    // cooperative A tile load: 64x64 f16 contiguous = 512 uint4
    {
        const uint4* src = (const uint4*)(A1 + (size_t)e0 * AKC);
        uint4* dst = (uint4*)As;
        dst[tid]       = src[tid];
        dst[tid + 256] = src[tid + 256];
    }
    __syncthreads();

    // layer-1 A fragments are invariant across all k-chunks: load once.
    const int r1 = w & 3;            // phase-1 row-tile owned by this wave
    const int cs = (w >> 2) * 2;     // phase-1 col-tile pair
    v16h a1f0 = load_a_frag(As + r1 * 16 * AKC + 0,  AKC, lane);
    v16h a1f1 = load_a_frag(As + r1 * 16 * AKC + 32, AKC, lane);

    v8f C[4][4];
    const v8f vzero = {0.f, 0.f, 0.f, 0.f, 0.f, 0.f, 0.f, 0.f};
#pragma unroll
    for (int r = 0; r < 4; ++r)
#pragma unroll
        for (int c = 0; c < 4; ++c) C[r][c] = vzero;

    for (int kc = 0; kc < HDIMC; kc += 64) {
        __syncthreads();  // previous phase-2 done reading Hs

        // phase 1: H chunk = gelu(A @ W1chunk + b1). 16 tiles, 2 per wave.
#pragma unroll
        for (int ci = 0; ci < 2; ++ci) {
            int ct  = cs + ci;
            int gct = (kc >> 4) + ct;                 // global col-tile 0..127
            v16h b0 = load_frag32(W1s, gct,       lane);   // k-block 0
            v16h b1 = load_frag32(W1s, 128 + gct, lane);   // k-block 1
            v8f acc = vzero;
            acc = __builtin_amdgcn_wmma_f32_16x16x32_f16(
                false, a1f0, false, b0, (short)0, acc, false, false);
            acc = __builtin_amdgcn_wmma_f32_16x16x32_f16(
                false, a1f1, false, b1, (short)0, acc, false, false);
            int col = ct * 16 + n;
            float bb = b1c[kc + col];
#pragma unroll
            for (int v = 0; v < 8; ++v) {
                float x = acc[v] + bb;
                Hs[(r1 * 16 + half * 8 + v) * 64 + col] = (_Float16)gelu_tanh(x);
            }
        }
        __syncthreads();

        // prefetch next chunk's W2 fragments for this wave's col-tiles
        if (kc + 64 < HDIMC) {
            int ktn = (kc + 64) >> 5;
            __builtin_prefetch(W2s + ((size_t)(ktn * 32 + w * 4) * 32 + lane) * 16, 0, 1);
            __builtin_prefetch(W2s + ((size_t)((ktn + 1) * 32 + w * 4) * 32 + lane) * 16, 0, 1);
        }

        // phase 2: C += H_chunk @ W2[kc:kc+64, :]. wave w owns col-tiles 4w..4w+3.
        v16h ah[4][2];
#pragma unroll
        for (int r = 0; r < 4; ++r) {
            ah[r][0] = load_a_frag(Hs + r * 16 * 64 + 0,  64, lane);
            ah[r][1] = load_a_frag(Hs + r * 16 * 64 + 32, 64, lane);
        }
        int kt = kc >> 5;
#pragma unroll
        for (int c = 0; c < 4; ++c) {
            int ct = w * 4 + c;
            v16h b0 = load_frag32(W2s, (kt)     * 32 + ct, lane);
            v16h b1 = load_frag32(W2s, (kt + 1) * 32 + ct, lane);
#pragma unroll
            for (int r = 0; r < 4; ++r) {
                C[r][c] = __builtin_amdgcn_wmma_f32_16x16x32_f16(
                    false, ah[r][0], false, b0, (short)0, C[r][c], false, false);
                C[r][c] = __builtin_amdgcn_wmma_f32_16x16x32_f16(
                    false, ah[r][1], false, b1, (short)0, C[r][c], false, false);
            }
        }
    }

    // epilogue: edge tokens = INV_SQRT3 * (C + rbf_b2 + dir_b2 + type_emb[1])
#pragma unroll
    for (int r = 0; r < 4; ++r) {
#pragma unroll
        for (int c = 0; c < 4; ++c) {
            int col = (w * 4 + c) * 16 + n;
            float add = badd[col];
#pragma unroll
            for (int v = 0; v < 8; ++v) {
                int row = e0 + r * 16 + half * 8 + v;
                out_tokens[(size_t)(N_NODESC + row) * EMBEDC + col] =
                    INV_SQRT3C * (C[r][c][v] + add);
            }
        }
    }
}

// ---- edges + one-hot masks ----------------------------------------------
__global__ void edges_masks_kernel(const int* __restrict__ ei,
                                   const int* __restrict__ dnd,
                                   const int* __restrict__ sns,
                                   float* __restrict__ out_edges,
                                   float* __restrict__ out_masks) {
    int j = blockIdx.x * blockDim.x + threadIdx.x;
    if (j >= TOTALEC) return;
    int r0, r1, t;
    if (j < N_EDGESC) {
        r0 = ei[j]; r1 = ei[N_EDGESC + j]; t = 0;
    } else if (j < 2 * N_EDGESC) {
        int e = j - N_EDGESC; r0 = ei[e]; r1 = N_NODESC + e; t = 1;
    } else if (j < 3 * N_EDGESC) {
        int e = j - 2 * N_EDGESC; r0 = N_NODESC + e; r1 = ei[N_EDGESC + e]; t = 2;
    } else if (j < 3 * N_EDGESC + PAIRSC) {
        int p = j - 3 * N_EDGESC; r0 = dnd[p]; r1 = dnd[PAIRSC + p]; t = 3;
    } else {
        int p = j - 3 * N_EDGESC - PAIRSC; r0 = sns[p]; r1 = sns[PAIRSC + p]; t = 4;
    }
    out_edges[j] = (float)r0;
    out_edges[(size_t)TOTALEC + j] = (float)r1;
#pragma unroll
    for (int t2 = 0; t2 < 5; ++t2)
        out_masks[(size_t)t2 * TOTALEC + j] = (t2 == t) ? 1.0f : 0.0f;
}

extern "C" void kernel_launch(void* const* d_in, const int* in_sizes, int n_in,
                              void* d_out, int out_size, void* d_ws, size_t ws_size,
                              hipStream_t stream) {
    const float* pos      = (const float*)d_in[0];
    const float* anum_emb = (const float*)d_in[1];
    const float* type_emb = (const float*)d_in[2];
    const float* rbf_w1   = (const float*)d_in[3];
    const float* rbf_b1   = (const float*)d_in[4];
    const float* rbf_w2   = (const float*)d_in[5];
    const float* rbf_b2   = (const float*)d_in[6];
    const float* dir_w1   = (const float*)d_in[7];
    const float* dir_b1   = (const float*)d_in[8];
    const float* dir_w2   = (const float*)d_in[9];
    const float* dir_b2   = (const float*)d_in[10];
    const int*   anum     = (const int*)d_in[11];
    const int*   ei       = (const int*)d_in[12];
    const int*   dnd      = (const int*)d_in[13];
    const int*   sns      = (const int*)d_in[14];

    // workspace layout (64B-aligned slabs)
    char* ws = (char*)d_ws;
    size_t off = 0;
    _Float16* A1  = (_Float16*)(ws + off); off += (size_t)N_EDGESC * AKC * 2;  // 30,720,000
    _Float16* W1s = (_Float16*)(ws + off); off += (size_t)AKC * HDIMC * 2;     //    262,144
    float*    b1c = (float*)   (ws + off); off += (size_t)HDIMC * 4;           //      8,192
    _Float16* W2s = (_Float16*)(ws + off); off += (size_t)HDIMC * EMBEDC * 2;  //  2,097,152
    float*    badd= (float*)   (ws + off); off += (size_t)EMBEDC * 4;

    float* out = (float*)d_out;
    float* out_tokens = out;                                                   // [(N+E), 512]
    float* out_edges  = out + (size_t)(N_NODESC + N_EDGESC) * EMBEDC;          // [2, T]
    float* out_masks  = out_edges + (size_t)2 * TOTALEC;                       // [5, T]

    const int B = 256;
    // weight prep (fragment-swizzled)
    prep_w1_swz     <<<(AKC * HDIMC + B - 1) / B, B, 0, stream>>>(rbf_w1, dir_w1, W1s);
    prep_b1_kernel  <<<(HDIMC + B - 1) / B, B, 0, stream>>>(rbf_b1, dir_b1, b1c);
    prep_w2_swz     <<<(HDIMC * EMBEDC + B - 1) / B, B, 0, stream>>>(rbf_w2, dir_w2, W2s);
    prep_badd_kernel<<<(EMBEDC + B - 1) / B, B, 0, stream>>>(rbf_b2, dir_b2, type_emb, badd);
    // node tokens
    node_tok_kernel<<<(N_NODESC * EMBEDC + B - 1) / B, B, 0, stream>>>(
        anum_emb, type_emb, anum, out_tokens);
    // edge features -> A matrix
    edge_geom_kernel<<<(N_EDGESC + B - 1) / B, B, 0, stream>>>(pos, ei, A1);
    // fused WMMA MLP: 240000/64 = 3750 blocks
    fused_edge_mlp<<<N_EDGESC / 64, 256, 0, stream>>>(A1, W1s, b1c, W2s, badd, out_tokens);
    // edges + masks
    edges_masks_kernel<<<(TOTALEC + B - 1) / B, B, 0, stream>>>(ei, dnd, sns, out_edges, out_masks);
    (void)in_sizes; (void)n_in; (void)out_size; (void)ws_size;
}
